// R_HGNN_Layer_31001073942569
// MI455X (gfx1250) — matrix operand order repaired and checked by reading, not compiled, not run on hardware
//
#include <hip/hip_runtime.h>

typedef __attribute__((ext_vector_type(16))) _Float16 v16h;
typedef __attribute__((ext_vector_type(8)))  _Float16 v8h;
typedef __attribute__((ext_vector_type(4)))  _Float16 v4h;
typedef __attribute__((ext_vector_type(2)))  _Float16 v2h;
typedef __attribute__((ext_vector_type(8)))  float    v8f;

constexpr int R = 3, N = 50000, E = 600000, HH = 8, DD = 16, HD = 128, RIN = 64;
constexpr int TOTROWS = R * N;                  // 150000
constexpr long NHD  = (long)TOTROWS * HD;       // 19,200,000
constexpr long NSEG = (long)TOTROWS * HH;       // 1,200,000

// ---------------- init: zero agg/denom, e_max = -inf ----------------
__global__ void init_kernel(float* __restrict__ agg, float* __restrict__ denom,
                            float* __restrict__ emax, long nAgg, long nSeg) {
  long t = (long)blockIdx.x * blockDim.x + threadIdx.x;
  if (t < nAgg) agg[t] = 0.0f;
  if (t < nSeg) { denom[t] = 0.0f; emax[t] = __int_as_float(0xFF800000); }
}

// ---------------- fused GEMM: h = feats@W_node ; res = feats@W_res + b_res ----
// blockIdx.y selects the weight matrix; 64 rows x 128 cols per block, 8 waves.
// sWsw holds the weight tile pre-swizzled into per-lane WMMA B-fragment order so
// each fragment is a single contiguous 32B LDS read (2x ds_load_b128).
__global__ __launch_bounds__(256)
void gemm_node_res_kernel(const float* __restrict__ feats,
                          const float* __restrict__ W_node,
                          const float* __restrict__ W_res,
                          const float* __restrict__ b_res,
                          float* __restrict__ hbuf,
                          float* __restrict__ resbuf) {
  __shared__ __attribute__((aligned(32))) _Float16 sA[64 * 128];     // 16 KB, row-major
  __shared__ __attribute__((aligned(32))) _Float16 sWsw[128 * 128];  // 32 KB, swizzled
  const int tid = threadIdx.x;
  const int matSel = blockIdx.y;
  const int rowBase = blockIdx.x * 64;
  const float* Wsrc = matSel ? W_res : W_node;

  // Weights: read (k,n),(k+1,n) coalesced over n; store packed pair at swizzled slot.
  // Swizzle: dst lane = (k&16 ? 16:0) + (n&15); within-lane idx = k&15;
  //          fragment slot = ((k>>5)*8 + (n>>4)).
  #pragma unroll 4
  for (int j = tid; j < 64 * 128; j += 256) {
    int kp = j >> 7;          // k pair index 0..63
    int n  = j & 127;
    int k  = kp * 2;
    float w0 = Wsrc[(size_t)k * 128 + n];
    float w1 = Wsrc[(size_t)(k + 1) * 128 + n];
    int lane = ((k & 16) ? 16 : 0) + (n & 15);
    int dst  = ((((k >> 5) * 8 + (n >> 4)) * 32 + lane) << 4) + (k & 15);
    *(v2h*)&sWsw[dst] = (v2h){(_Float16)w0, (_Float16)w1};
  }
  // A tile: float4 coalesced reads -> v4h 8B LDS stores, row-major.
  #pragma unroll 2
  for (int j = tid; j < 64 * 32; j += 256) {   // 2048 float4 chunks
    int row = j >> 5;
    int c4  = (j & 31) * 4;
    int gr  = rowBase + row;
    float4 f = (gr < TOTROWS) ? *(const float4*)&feats[(size_t)gr * 128 + c4]
                              : make_float4(0.f, 0.f, 0.f, 0.f);
    *(v4h*)&sA[row * 128 + c4] =
        (v4h){(_Float16)f.x, (_Float16)f.y, (_Float16)f.z, (_Float16)f.w};
  }
  __syncthreads();

  const int w = tid >> 5, lane = tid & 31;
  const int rowTile = w & 3;       // which 16-row tile
  const int colHalf = w >> 2;      // cols 0..63 or 64..127
  const int half = lane >> 4;      // lane group
  const int mrow = lane & 15;

  v8f acc[4] = {};
  const _Float16* aRow = &sA[(rowTile * 16 + mrow) * 128];

  #pragma unroll
  for (int kk = 0; kk < 128; kk += 32) {
    // A fragment: one contiguous 16-half run, k = 16*half + s (matches B swizzle).
    v16h afrag = *(const v16h*)(aRow + kk + half * 16);
    #pragma unroll
    for (int t = 0; t < 4; ++t) {
      int ct = colHalf * 4 + t;
      v16h bfrag = *(const v16h*)&sWsw[((((kk >> 5) * 8 + ct) * 32 + lane) << 4)];
      acc[t] = __builtin_amdgcn_wmma_f32_16x16x32_f16(
          false, afrag, false, bfrag, (short)0, acc[t], false, false);
    }
  }

  float* out = matSel ? resbuf : hbuf;
  #pragma unroll
  for (int t = 0; t < 4; ++t) {
    int col = (colHalf * 4 + t) * 16 + mrow;
    float bias = matSel ? b_res[col] : 0.0f;
    #pragma unroll
    for (int j = 0; j < 8; ++j) {
      int row = rowBase + rowTile * 16 + half * 8 + j;   // C layout: M=j(+8)
      if (row < TOTROWS) out[(size_t)row * 128 + col] = acc[t][j] + bias;
    }
  }
}

// ---------------- rel_att = rel_emb @ W_rel  (R x 256) ----------------
__global__ void relatt_kernel(const float* __restrict__ rel_emb,
                              const float* __restrict__ W_rel,
                              float* __restrict__ ratt) {
  int r = blockIdx.x, o = threadIdx.x;  // 256 threads
  float s = 0.0f;
  #pragma unroll 8
  for (int i = 0; i < RIN; ++i)
    s += rel_emb[r * RIN + i] * W_rel[((size_t)r * RIN + i) * 256 + o];
  ratt[r * 256 + o] = s;
}

// ---------------- per-node logits e_src/e_dst ----------------
__global__ void node_att_kernel(const float* __restrict__ hbuf,
                                const float* __restrict__ ratt,
                                float* __restrict__ esrc, float* __restrict__ edstv) {
  int t = blockIdx.x * blockDim.x + threadIdx.x;
  if (t >= R * N * HH) return;
  int h = t & 7, rn = t >> 3, r = rn / N;
  const float* hp = hbuf + (size_t)rn * 128 + h * DD;
  const float* ap = ratt + r * 256 + h * 32;
  float s1 = 0.0f, s2 = 0.0f;
  #pragma unroll
  for (int d = 0; d < DD; ++d) {
    float hv = hp[d];
    s1 += hv * ap[d];
    s2 += hv * ap[16 + d];
  }
  esrc[t] = s1;
  edstv[t] = s2;
}

__device__ __forceinline__ float lrelu(float v) { return v > 0.0f ? v : 0.2f * v; }

// ---------------- edge pass 1: segment max (monotonic-bits atomics) ----------
__global__ void edge_max_kernel(const int* __restrict__ src, const int* __restrict__ dst,
                                const float* __restrict__ esrc, const float* __restrict__ edstv,
                                float* __restrict__ emax) {
  int t = blockIdx.x * blockDim.x + threadIdx.x;
  if (t >= R * E) return;
  int r = t / E;
  int s = src[t], d = dst[t];
  const float* ps = esrc + ((size_t)r * N + s) * HH;
  const float* pd = edstv + ((size_t)r * N + d) * HH;
  float* pm = emax + ((size_t)r * N + d) * HH;
  #pragma unroll
  for (int h = 0; h < HH; ++h) {
    float v = lrelu(ps[h] + pd[h]);
    int iv = __float_as_int(v);
    if (iv >= 0)
      __hip_atomic_fetch_max((int*)(pm + h), iv,
                             __ATOMIC_RELAXED, __HIP_MEMORY_SCOPE_AGENT);
    else
      __hip_atomic_fetch_min((unsigned int*)(pm + h), (unsigned int)iv,
                             __ATOMIC_RELAXED, __HIP_MEMORY_SCOPE_AGENT);
  }
}

// ---------------- edge pass 2: denom = segment_sum(exp(e - max)) ------------
__global__ void edge_sum_kernel(const int* __restrict__ src, const int* __restrict__ dst,
                                const float* __restrict__ esrc, const float* __restrict__ edstv,
                                const float* __restrict__ emax, float* __restrict__ denom) {
  int t = blockIdx.x * blockDim.x + threadIdx.x;
  if (t >= R * E) return;
  int r = t / E;
  int s = src[t], d = dst[t];
  const float* ps = esrc + ((size_t)r * N + s) * HH;
  const float* pd = edstv + ((size_t)r * N + d) * HH;
  const float* pm = emax + ((size_t)r * N + d) * HH;
  float* pn = denom + ((size_t)r * N + d) * HH;
  #pragma unroll
  for (int h = 0; h < HH; ++h) {
    float ex = __expf(lrelu(ps[h] + pd[h]) - pm[h]);
    __hip_atomic_fetch_add(pn + h, ex, __ATOMIC_RELAXED, __HIP_MEMORY_SCOPE_AGENT);
  }
}

// ---------------- edge pass 3: agg += h[src] * a  (per edge,head) -----------
__global__ void edge_agg_kernel(const int* __restrict__ src, const int* __restrict__ dst,
                                const float* __restrict__ esrc, const float* __restrict__ edstv,
                                const float* __restrict__ emax, const float* __restrict__ denom,
                                const float* __restrict__ hbuf, float* __restrict__ agg) {
  int t = blockIdx.x * blockDim.x + threadIdx.x;
  if (t >= R * E * HH) return;
  int h = t & 7, te = t >> 3;
  int r = te / E;
  int s = src[te], d = dst[te];
  size_t ns = (size_t)r * N + s, nd = (size_t)r * N + d;
  float v = lrelu(esrc[ns * HH + h] + edstv[nd * HH + h]);
  float ex = __expf(v - emax[nd * HH + h]);
  float a = ex / fmaxf(denom[nd * HH + h], 1e-16f);
  const float4* hp = (const float4*)(hbuf + ns * 128 + h * DD);
  float* ap = agg + nd * 128 + h * DD;
  #pragma unroll
  for (int q = 0; q < 4; ++q) {
    float4 hv = hp[q];
    __hip_atomic_fetch_add(ap + 4 * q + 0, hv.x * a, __ATOMIC_RELAXED, __HIP_MEMORY_SCOPE_AGENT);
    __hip_atomic_fetch_add(ap + 4 * q + 1, hv.y * a, __ATOMIC_RELAXED, __HIP_MEMORY_SCOPE_AGENT);
    __hip_atomic_fetch_add(ap + 4 * q + 2, hv.z * a, __ATOMIC_RELAXED, __HIP_MEMORY_SCOPE_AGENT);
    __hip_atomic_fetch_add(ap + 4 * q + 3, hv.w * a, __ATOMIC_RELAXED, __HIP_MEMORY_SCOPE_AGENT);
  }
}

// ---------------- residual mix: agg = relu(agg)*a + res*(1-a) ---------------
__global__ void mix_kernel(float* __restrict__ agg, const float* __restrict__ res,
                           const float* __restrict__ res_w) {
  long t = (long)blockIdx.x * blockDim.x + threadIdx.x;
  if (t >= NHD) return;
  float alpha = 1.0f / (1.0f + __expf(-res_w[0]));
  float v = agg[t];
  v = v > 0.0f ? v : 0.0f;
  agg[t] = v * alpha + res[t] * (1.0f - alpha);
}

// ---------------- cross-relation softmax over R ----------------
__global__ void cross_kernel(const float* __restrict__ mix, const float* __restrict__ cross_w,
                             float* __restrict__ node_out) {
  int t = blockIdx.x * blockDim.x + threadIdx.x;
  if (t >= R * N * HH) return;
  int h = t & 7, tn = t >> 3;
  int rp = tn / N, n = tn - rp * N;
  float wv[DD];
  const float* wp = cross_w + (rp * HH + h) * DD;
  #pragma unroll
  for (int d = 0; d < DD; ++d) wv[d] = wp[d];
  float x[3][DD], sc[3];
  #pragma unroll
  for (int r = 0; r < 3; ++r) {
    const float* xp = mix + ((size_t)r * N + n) * 128 + h * DD;
    float s = 0.0f;
    #pragma unroll
    for (int d = 0; d < DD; ++d) { x[r][d] = xp[d]; s += x[r][d] * wv[d]; }
    sc[r] = lrelu(s);
  }
  float m = fmaxf(sc[0], fmaxf(sc[1], sc[2]));
  float e0 = __expf(sc[0] - m), e1 = __expf(sc[1] - m), e2 = __expf(sc[2] - m);
  float inv = 1.0f / (e0 + e1 + e2);
  float a0 = e0 * inv, a1 = e1 * inv, a2 = e2 * inv;
  float* op = node_out + ((size_t)rp * N + n) * 128 + h * DD;
  #pragma unroll
  for (int d = 0; d < DD; ++d)
    op[d] = x[0][d] * a0 + x[1][d] * a1 + x[2][d] * a2;
}

// ---------------- rel_out = rel_emb @ W_prop + b_prop ----------------
__global__ void relout_kernel(const float* __restrict__ rel_emb, const float* __restrict__ W_prop,
                              const float* __restrict__ b_prop, float* __restrict__ out2) {
  int r = blockIdx.x, o = threadIdx.x;  // 128 threads
  float s = b_prop[r * 128 + o];
  #pragma unroll 8
  for (int i = 0; i < RIN; ++i)
    s += rel_emb[r * RIN + i] * W_prop[((size_t)r * RIN + i) * 128 + o];
  out2[r * 128 + o] = s;
}

extern "C" void kernel_launch(void* const* d_in, const int* in_sizes, int n_in,
                              void* d_out, int out_size, void* d_ws, size_t ws_size,
                              hipStream_t stream) {
  const float* feats   = (const float*)d_in[0];
  const float* rel_emb = (const float*)d_in[1];
  const int*   src     = (const int*)d_in[2];
  const int*   dst     = (const int*)d_in[3];
  const float* W_node  = (const float*)d_in[4];
  const float* W_rel   = (const float*)d_in[5];
  const float* W_res   = (const float*)d_in[6];
  const float* b_res   = (const float*)d_in[7];
  const float* res_w   = (const float*)d_in[8];
  const float* cross_w = (const float*)d_in[9];
  const float* W_prop  = (const float*)d_in[10];
  const float* b_prop  = (const float*)d_in[11];

  float* node_out = (float*)d_out;
  float* rel_out  = node_out + (size_t)TOTROWS * HD;

  float* hbuf   = (float*)d_ws;
  float* resbuf = hbuf   + NHD;
  float* aggbuf = resbuf + NHD;
  float* esrc   = aggbuf + NHD;
  float* edstv  = esrc   + NSEG;
  float* emax   = edstv  + NSEG;
  float* denom  = emax   + NSEG;
  float* rattbf = denom  + NSEG;

  init_kernel<<<(int)((NHD + 255) / 256), 256, 0, stream>>>(aggbuf, denom, emax, NHD, NSEG);

  dim3 g1((TOTROWS + 63) / 64, 2);
  gemm_node_res_kernel<<<g1, 256, 0, stream>>>(feats, W_node, W_res, b_res, hbuf, resbuf);

  relatt_kernel<<<R, 256, 0, stream>>>(rel_emb, W_rel, rattbf);
  node_att_kernel<<<(R * N * HH + 255) / 256, 256, 0, stream>>>(hbuf, rattbf, esrc, edstv);

  const int edges = R * E;
  edge_max_kernel<<<(edges + 255) / 256, 256, 0, stream>>>(src, dst, esrc, edstv, emax);
  edge_sum_kernel<<<(edges + 255) / 256, 256, 0, stream>>>(src, dst, esrc, edstv, emax, denom);
  edge_agg_kernel<<<(edges * HH + 255) / 256, 256, 0, stream>>>(src, dst, esrc, edstv, emax,
                                                                denom, hbuf, aggbuf);

  mix_kernel<<<(int)((NHD + 255) / 256), 256, 0, stream>>>(aggbuf, resbuf, res_w);
  cross_kernel<<<(R * N * HH + 255) / 256, 256, 0, stream>>>(aggbuf, cross_w, node_out);
  relout_kernel<<<R, 128, 0, stream>>>(rel_emb, W_prop, b_prop, rel_out);
}
